// FlaxWav2Vec2GumbelVectorQuantizer_55327768707929
// MI455X (gfx1250) — matrix operand order repaired and checked by reading, not compiled, not run on hardware
//
#include <hip/hip_runtime.h>
#include <hip/hip_bf16.h>

typedef __attribute__((ext_vector_type(16))) __bf16 v16bf;
typedef __attribute__((ext_vector_type(8)))  __bf16 v8bf;
typedef __attribute__((ext_vector_type(8)))  float  v8f;
typedef __attribute__((ext_vector_type(4)))  float  f32x4;

#define ROWS 32          // rows (B*T) per block
#define HDIM 512         // hidden dim (K of GEMM1)
#define NCOL 640         // G*V (N of GEMM1)
#define VPG  320         // vars per group (K of GEMM2)
#define DG   128         // D/G (N of GEMM2 per group)
#define NT   256         // threads per block (8 waves)

union FragBF { v16bf v; v8bf h[2]; };

// ---------------- prep: bf16 transposed weights + zero marginal ----------------
__global__ __launch_bounds__(256) void vq_prep_kernel(
    const float* __restrict__ pk,   // proj_kernel [512,640]
    const float* __restrict__ cb,   // codevectors [640,128]
    __bf16* __restrict__ BT,        // out [640,512]  (N-major, K contiguous)
    __bf16* __restrict__ cbT,       // out [(g*128+d)*320+v]
    float*  __restrict__ marg)      // out [640] zeroed
{
  int idx = blockIdx.x * 256 + threadIdx.x;
  if (idx < NCOL * HDIM) {
    int n = idx / HDIM, k = idx % HDIM;
    BT[idx] = (__bf16)pk[k * NCOL + n];
  } else if (idx < NCOL * HDIM + 2 * DG * VPG) {
    int j = idx - NCOL * HDIM;
    int g = j / (DG * VPG);
    int r = j % (DG * VPG);
    int d = r / VPG, v = r % VPG;
    cbT[j] = (__bf16)cb[(g * VPG + v) * DG + d];
  } else if (idx < NCOL * HDIM + 2 * DG * VPG + NCOL) {
    marg[idx - NCOL * HDIM - 2 * DG * VPG] = 0.f;
  }
}

// ---------------- fused VQ: proj GEMM -> dual softmax -> codevector GEMM ----------------
__global__ __launch_bounds__(256) void vq_main_kernel(
    const float* __restrict__ hs,          // [65536,512]
    const float* __restrict__ gum,         // [131072,320]
    const float* __restrict__ bias,        // [640]
    const unsigned char* __restrict__ mask,// [65536] bool
    const __bf16* __restrict__ BT,         // [640,512] bf16
    const __bf16* __restrict__ cbT,        // [256,320] bf16
    float* __restrict__ marg,              // [640] global accumulator
    float* __restrict__ out)               // [65536,256]
{
  __shared__ float sLogits[ROWS * NCOL];                    // 80 KB
  __shared__ __align__(16) __bf16 sAP[ROWS * NCOL];         // 40 KB: A-tile then probs
  __shared__ __align__(16) float sGum[ROWS * 2 * VPG];      // 80 KB gumbel slab
  __shared__ float sMarg[NCOL];                             // 2.5 KB

  const int tid  = threadIdx.x;
  const int wave = tid >> 5;
  const int lane = tid & 31;
  const int h    = lane >> 4;   // lane half selects K sub-range
  const int lm   = lane & 15;   // row (A) / col (B,C,D) within tile
  const int rowBase = blockIdx.x * ROWS;

  // ---- kick off async DMA of this block's gumbel slab (80 KB) into LDS.
  //      It runs concurrently with A-staging and the pass-1 WMMA GEMM.
  {
    unsigned ldsAddr = (unsigned)(size_t)(&sGum[0]) + tid * 16u;
    const char* gsrc = (const char*)gum + (size_t)rowBase * 2 * VPG * 4 + tid * 16;
#pragma unroll
    for (int i = 0; i < (ROWS * 2 * VPG * 4) / (NT * 16); ++i) {   // 20 chunks
      asm volatile("global_load_async_to_lds_b128 %0, %1, off"
                   :: "v"(ldsAddr + i * NT * 16u), "v"(gsrc + i * NT * 16)
                   : "memory");
    }
  }

  for (int i = tid; i < NCOL; i += NT) sMarg[i] = 0.f;

  // ---- stage A tile: fp32 global -> bf16 LDS (32 x 512) ----
  {
    const f32x4* src = (const f32x4*)(hs + (size_t)rowBase * HDIM);
    for (int i = tid; i < ROWS * HDIM / 4; i += NT) {
      f32x4 f = src[i];
      __bf16* dst = &sAP[i * 4];
      dst[0] = (__bf16)f.x; dst[1] = (__bf16)f.y;
      dst[2] = (__bf16)f.z; dst[3] = (__bf16)f.w;
    }
  }
  __syncthreads();

  // ---- pass 1: logits = A @ W + bias   (2 rt x 40 ct tiles, 10 per wave) ----
  for (int i = 0; i < 10; ++i) {
    int t  = wave * 10 + i;
    int ct = t % 40, rt = t / 40;
    const __bf16* arow = &sAP[(rt * 16 + lm) * HDIM];
    const __bf16* brow = BT + (size_t)(ct * 16 + lm) * HDIM;
    v8f acc = {};
#pragma unroll 4
    for (int ks = 0; ks < HDIM / 32; ++ks) {
      FragBF a, b;
      a.h[0] = *(const v8bf*)(arow + ks * 32 + h * 8);
      a.h[1] = *(const v8bf*)(arow + ks * 32 + 16 + h * 8);
      b.h[0] = *(const v8bf*)(brow + ks * 32 + h * 8);
      b.h[1] = *(const v8bf*)(brow + ks * 32 + 16 + h * 8);
      acc = __builtin_amdgcn_wmma_f32_16x16x32_bf16(false, a.v, false, b.v,
                                                    (short)0, acc, false, false);
    }
    int col = ct * 16 + lm;
    float bv = bias[col];
#pragma unroll
    for (int r = 0; r < 8; ++r)
      sLogits[(rt * 16 + h * 8 + r) * NCOL + col] = acc[r] + bv;
  }

  // gumbel DMA must be complete and visible to all waves before softmax
  asm volatile("s_wait_asynccnt 0x0" ::: "memory");
  __syncthreads();

  // ---- pass 2: dual softmax per (row, group); 4 lanes per (row,g) team ----
  {
    const int team = tid >> 2;       // 0..63
    const int sub  = tid & 3;
    const int row  = team >> 1;
    const int g    = team & 1;
    const float* lrow = &sLogits[row * NCOL + g * VPG];
    const float* grow = &sGum[(row * 2 + g) * VPG];

    float m1 = -3.0e38f, m2 = -3.0e38f;
    for (int v = sub; v < VPG; v += 4) {
      float l = lrow[v];
      float y = (l + grow[v]) * 0.5f;     // (logit+gumbel)/TEMP, TEMP=2
      m1 = fmaxf(m1, y);
      m2 = fmaxf(m2, l);
    }
    m1 = fmaxf(m1, __shfl_xor(m1, 1, 32)); m1 = fmaxf(m1, __shfl_xor(m1, 2, 32));
    m2 = fmaxf(m2, __shfl_xor(m2, 1, 32)); m2 = fmaxf(m2, __shfl_xor(m2, 2, 32));

    float s1 = 0.f, s2 = 0.f;
    for (int v = sub; v < VPG; v += 4) {
      float l = lrow[v];
      s1 += __expf((l + grow[v]) * 0.5f - m1);
      s2 += __expf(l - m2);
    }
    s1 += __shfl_xor(s1, 1, 32); s1 += __shfl_xor(s1, 2, 32);
    s2 += __shfl_xor(s2, 1, 32); s2 += __shfl_xor(s2, 2, 32);

    float inv1 = 1.f / s1;
    float mv   = mask[rowBase + row] ? 1.f : 0.f;
    float inv2 = mv / s2;
    __bf16* prow = &sAP[row * NCOL + g * VPG];  // A-tile is dead; reuse as probs
    for (int v = sub; v < VPG; v += 4) {
      float l = lrow[v];
      prow[v] = (__bf16)(__expf((l + grow[v]) * 0.5f - m1) * inv1);
      atomicAdd(&sMarg[g * VPG + v], __expf(l - m2) * inv2);
    }
  }
  __syncthreads();

  for (int i = tid; i < NCOL; i += NT) atomicAdd(&marg[i], sMarg[i]);

  // ---- pass 3: cv = probs @ cb   (per group: [32x320] @ [320x128]) ----
  for (int i = 0; i < 4; ++i) {
    int t   = wave * 4 + i;          // 0..31
    int g   = t >> 4;
    int rem = t & 15;
    int rt  = rem >> 3, ct = rem & 7;
    const __bf16* arow = &sAP[(rt * 16 + lm) * NCOL + g * VPG];
    const __bf16* brow = cbT + (size_t)(g * DG + ct * 16 + lm) * VPG;
    v8f acc = {};
#pragma unroll
    for (int ks = 0; ks < VPG / 32; ++ks) {
      FragBF a, b;
      a.h[0] = *(const v8bf*)(arow + ks * 32 + h * 8);
      a.h[1] = *(const v8bf*)(arow + ks * 32 + 16 + h * 8);
      b.h[0] = *(const v8bf*)(brow + ks * 32 + h * 8);
      b.h[1] = *(const v8bf*)(brow + ks * 32 + 16 + h * 8);
      acc = __builtin_amdgcn_wmma_f32_16x16x32_bf16(false, a.v, false, b.v,
                                                    (short)0, acc, false, false);
    }
    int col = g * DG + ct * 16 + lm;
#pragma unroll
    for (int r = 0; r < 8; ++r) {
      int row = rowBase + rt * 16 + h * 8 + r;
      out[(size_t)row * 256 + col] = acc[r];
    }
  }
}

// ---------------- finalize: perplexity scalar ----------------
__global__ __launch_bounds__(256) void vq_perp_kernel(
    const unsigned char* __restrict__ mask,
    const float* __restrict__ marg,
    float* __restrict__ outp, int nrows)
{
  __shared__ float sSum;
  __shared__ float sG[2];
  int tid = threadIdx.x;
  if (tid == 0) sSum = 0.f;
  if (tid < 2) sG[tid] = 0.f;
  __syncthreads();

  float local = 0.f;
  for (int i = tid; i < nrows; i += 256) local += mask[i] ? 1.f : 0.f;
  atomicAdd(&sSum, local);
  __syncthreads();

  float inv = 1.f / sSum;
  float a0 = 0.f, a1 = 0.f;
  for (int i = tid; i < NCOL; i += 256) {
    float m = marg[i] * inv;
    float t = m * __logf(m + 1e-7f);
    if (i < VPG) a0 += t; else a1 += t;
  }
  atomicAdd(&sG[0], a0);
  atomicAdd(&sG[1], a1);
  __syncthreads();
  if (tid == 0) outp[0] = __expf(-sG[0]) + __expf(-sG[1]);
}

extern "C" void kernel_launch(void* const* d_in, const int* in_sizes, int n_in,
                              void* d_out, int out_size, void* d_ws, size_t ws_size,
                              hipStream_t stream)
{
  const float* hs   = (const float*)d_in[0];          // hidden_states
  const float* gum  = (const float*)d_in[1];          // gumbels
  const float* pk   = (const float*)d_in[2];          // proj_kernel
  const float* pb   = (const float*)d_in[3];          // proj_bias
  const float* cb   = (const float*)d_in[4];          // codevectors
  const unsigned char* mask = (const unsigned char*)d_in[5]; // bool mask

  const int nrows = in_sizes[0] / HDIM;               // B*T = 65536

  char* ws = (char*)d_ws;
  __bf16* BT   = (__bf16*)ws;                                   // 640*512*2  B
  __bf16* cbT  = (__bf16*)(ws + (size_t)NCOL * HDIM * 2);       // 2*128*320*2 B
  float*  marg = (float*)(ws + (size_t)NCOL * HDIM * 2 + (size_t)2 * DG * VPG * 2);

  float* out = (float*)d_out;

  const int prepElems = NCOL * HDIM + 2 * DG * VPG + NCOL;
  vq_prep_kernel<<<(prepElems + 255) / 256, 256, 0, stream>>>(pk, cb, BT, cbT, marg);
  vq_main_kernel<<<nrows / ROWS, NT, 0, stream>>>(hs, gum, pb, mask, BT, cbT, marg, out);
  vq_perp_kernel<<<1, 256, 0, stream>>>(mask, marg, out + (size_t)nrows * 256, nrows);
}
